// OLMoE1B7B0924Synapse_22514218566327
// MI455X (gfx1250) — compile-verified, matrix operand order checked by reading, not run on hardware
//
#include <hip/hip_runtime.h>
#include <math.h>

typedef __attribute__((ext_vector_type(16))) __bf16 v16bf;
typedef __attribute__((ext_vector_type(8)))  float  v8f;

#define SEQ    1024
#define DMODEL 2048
#define NHEAD  16
#define HDIM   128
#define NEXP   64
#define TOPK   8
#define MDIM   512
#define VOCAB  50304
#define EPS_   1e-5f
#define SCALE_ 0.08838834764831845f

#define BM 128
#define BN 128
#define BK 32
#define LDKP 40   // padded LDS row stride (bf16 elems): 80B rows, 16B-aligned slots

__device__ __forceinline__ v8f wmma_bf16(v16bf a, v16bf b, v8f c) {
  return __builtin_amdgcn_wmma_f32_16x16x32_bf16(false, a, false, b, (short)0, c,
                                                 false, false);
}

// pack two fp32 -> one dword of two bf16 (lowers to v_cvt_pk_bf16_f32)
__device__ __forceinline__ unsigned pk2(float a, float b) {
  union { __bf16 h[2]; unsigned u; } c;
  c.h[0] = (__bf16)a; c.h[1] = (__bf16)b;
  return c.u;
}
__device__ __forceinline__ uint4 pack8(float4 x, float4 y) {
  uint4 r;
  r.x = pk2(x.x, x.y); r.y = pk2(x.z, x.w);
  r.z = pk2(y.x, y.y); r.w = pk2(y.z, y.w);
  return r;
}

// Load a 16x32 bf16 fragment from LDS following the CDNA5 wave32 A/B layout:
// lane -> row (M or N), VGPR i (=vector halves 2i,2i+1) -> K = g*16 + (lane>=16?8:0) + 2j
__device__ __forceinline__ v16bf load_frag(const __bf16* base, int stride, int lane) {
  v16bf f;
  int row = lane & 15;
  int kb  = (lane < 16) ? 0 : 8;
#pragma unroll
  for (int g = 0; g < 2; ++g) {
#pragma unroll
    for (int j = 0; j < 4; ++j) {
      int k = g * 16 + kb + 2 * j;
      int i = g * 4 + j;
      f[2 * i]     = base[row * stride + k];
      f[2 * i + 1] = base[row * stride + k + 1];
    }
  }
  return f;
}

// ---------------------------------------------------------------------------
// Generic WMMA GEMM: C[M,N] = A[M,K] * B[N,K]^T  (fp32 in, bf16 WMMA, fp32 out)
//   residual   : optional, added at epilogue (same layout as C)
//   a_row_idx  : optional gather of A rows (MoE token gather)
//   c_row_idx  : optional scatter of C rows with atomicAdd (MoE down-proj)
//   expert_off : optional per-expert row ranges (blockIdx.z = expert)
// ---------------------------------------------------------------------------
__global__ __launch_bounds__(256)
void gemm_wmma_kernel(const float* __restrict__ A, int lda,
                      const float* __restrict__ B, int ldb,
                      float* __restrict__ C, int ldc,
                      int M, int N, int Kd,
                      const float* __restrict__ residual,
                      const int* __restrict__ a_row_idx,
                      const int* __restrict__ c_row_idx,
                      const int* __restrict__ expert_off,
                      long long b_expert_stride)
{
  __shared__ __align__(16) __bf16 As[BM * LDKP];
  __shared__ __align__(16) __bf16 Bs[BN * LDKP];

  int rowStart = 0, rowEnd = M;
  const float* Bp = B;
  if (expert_off) {
    int e    = blockIdx.z;
    rowStart = expert_off[e];
    rowEnd   = expert_off[e + 1];
    Bp       = B + (size_t)e * (size_t)b_expert_stride;
    if ((int)(blockIdx.y * BM) >= rowEnd - rowStart) return;  // uniform exit
  }

  int tid  = threadIdx.x;
  int sr   = tid >> 1;            // staging row 0..127
  int sk   = (tid & 1) * 16;      // staging k offset 0/16
  int wave = tid >> 5;
  int lane = tid & 31;
  int wm   = wave & 3;            // 4 waves along M (32 rows each)
  int wn   = wave >> 2;           // 2 waves along N (64 cols each)

  v8f acc[2][4];
#pragma unroll
  for (int s = 0; s < 2; ++s)
#pragma unroll
    for (int t = 0; t < 4; ++t)
#pragma unroll
      for (int r = 0; r < 8; ++r) acc[s][t][r] = 0.0f;

  int  growA = rowStart + blockIdx.y * BM + sr;
  bool aok   = growA < rowEnd;
  int  arow  = aok ? (a_row_idx ? a_row_idx[growA] : growA) : 0;
  int  gn    = blockIdx.x * BN + sr;
  bool bok   = gn < N;
  int  brow  = bok ? gn : 0;

  // 16B-aligned LDS slots for this thread (LDKP=40 => 80B rows; sk*2 in {0,32})
  uint4* asv = (uint4*)(As + sr * LDKP + sk);
  uint4* bsv = (uint4*)(Bs + sr * LDKP + sk);

  // zero-fill once: rows beyond range stay zero for every K step
  {
    uint4 z; z.x = z.y = z.z = z.w = 0u;
    asv[0] = z; asv[1] = z;
    bsv[0] = z; bsv[1] = z;
  }

  const float* arowp = A  + (size_t)arow * lda;
  const float* browp = Bp + (size_t)brow * ldb;

  for (int k0 = 0; k0 < Kd; k0 += BK) {
    if (aok) {
      const float4* ap4 = (const float4*)(arowp + k0 + sk);
      float4 f0 = ap4[0], f1 = ap4[1], f2 = ap4[2], f3 = ap4[3];
      asv[0] = pack8(f0, f1);
      asv[1] = pack8(f2, f3);
      if (k0 + BK < Kd) __builtin_prefetch(arowp + k0 + BK + sk, 0, 1);
    }
    if (bok) {
      const float4* bp4 = (const float4*)(browp + k0 + sk);
      float4 f0 = bp4[0], f1 = bp4[1], f2 = bp4[2], f3 = bp4[3];
      bsv[0] = pack8(f0, f1);
      bsv[1] = pack8(f2, f3);
      if (k0 + BK < Kd) __builtin_prefetch(browp + k0 + BK + sk, 0, 1);
    }
    __syncthreads();

    v16bf af[2], bf[4];
#pragma unroll
    for (int s = 0; s < 2; ++s)
      af[s] = load_frag(As + (wm * 32 + s * 16) * LDKP, LDKP, lane);
#pragma unroll
    for (int t = 0; t < 4; ++t)
      bf[t] = load_frag(Bs + (wn * 64 + t * 16) * LDKP, LDKP, lane);
#pragma unroll
    for (int s = 0; s < 2; ++s)
#pragma unroll
      for (int t = 0; t < 4; ++t)
        acc[s][t] = wmma_bf16(af[s], bf[t], acc[s][t]);
    __syncthreads();
  }

  // Epilogue: C layout VGPR r -> M = r (lanes 0-15) / r+8 (lanes 16-31), N = lane%16
#pragma unroll
  for (int s = 0; s < 2; ++s) {
#pragma unroll
    for (int t = 0; t < 4; ++t) {
#pragma unroll
      for (int r = 0; r < 8; ++r) {
        int lrow  = wm * 32 + s * 16 + ((lane < 16) ? r : r + 8);
        int grow2 = rowStart + blockIdx.y * BM + lrow;
        int gcol  = blockIdx.x * BN + wn * 64 + t * 16 + (lane & 15);
        if (grow2 < rowEnd && gcol < N) {
          float val = acc[s][t][r];
          if (c_row_idx) {
            atomicAdd(&C[(size_t)c_row_idx[grow2] * ldc + gcol], val);
          } else {
            size_t off = (size_t)grow2 * ldc + gcol;
            if (residual) val += residual[off];
            C[off] = val;
          }
        }
      }
    }
  }
}

// ---------------------------------------------------------------------------
// Embedding gather + RMSNorm (block per token)
// ---------------------------------------------------------------------------
__global__ __launch_bounds__(256)
void embed_rms_kernel(const int* __restrict__ ids, const float* __restrict__ embed,
                      const float* __restrict__ w, float* __restrict__ x,
                      float* __restrict__ xn)
{
  int t = blockIdx.x;
  __shared__ float red[256];
  const float* row = embed + (size_t)ids[t] * DMODEL;
  size_t base = (size_t)t * DMODEL;
  float ss = 0.f;
  for (int d = threadIdx.x; d < DMODEL; d += 256) {
    float v = row[d];
    x[base + d] = v;
    ss += v * v;
  }
  red[threadIdx.x] = ss;
  __syncthreads();
  for (int s = 128; s > 0; s >>= 1) {
    if (threadIdx.x < s) red[threadIdx.x] += red[threadIdx.x + s];
    __syncthreads();
  }
  float inv = rsqrtf(red[0] / (float)DMODEL + EPS_);
  for (int d = threadIdx.x; d < DMODEL; d += 256)
    xn[base + d] = x[base + d] * inv * w[d];
}

__global__ __launch_bounds__(256)
void rms_kernel(const float* __restrict__ in, const float* __restrict__ w,
                float* __restrict__ out)
{
  int t = blockIdx.x;
  __shared__ float red[256];
  size_t base = (size_t)t * DMODEL;
  float ss = 0.f;
  for (int d = threadIdx.x; d < DMODEL; d += 256) {
    float v = in[base + d];
    ss += v * v;
  }
  red[threadIdx.x] = ss;
  __syncthreads();
  for (int s = 128; s > 0; s >>= 1) {
    if (threadIdx.x < s) red[threadIdx.x] += red[threadIdx.x + s];
    __syncthreads();
  }
  float inv = rsqrtf(red[0] / (float)DMODEL + EPS_);
  for (int d = threadIdx.x; d < DMODEL; d += 256)
    out[base + d] = in[base + d] * inv * w[d];
}

// ---------------------------------------------------------------------------
// Fused q/k RMSNorm + RoPE (block per token, in-place)
// ---------------------------------------------------------------------------
__global__ __launch_bounds__(256)
void qk_rms_rope_kernel(float* __restrict__ q, float* __restrict__ k,
                        const float* __restrict__ qw, const float* __restrict__ kw)
{
  int t = blockIdx.x;
  __shared__ float rq[256], rk[256];
  size_t base = (size_t)t * DMODEL;
  float sq = 0.f, sk = 0.f;
  for (int d = threadIdx.x; d < DMODEL; d += 256) {
    float a = q[base + d]; sq += a * a;
    float b = k[base + d]; sk += b * b;
  }
  rq[threadIdx.x] = sq; rk[threadIdx.x] = sk;
  __syncthreads();
  for (int s = 128; s > 0; s >>= 1) {
    if (threadIdx.x < s) { rq[threadIdx.x] += rq[threadIdx.x + s];
                           rk[threadIdx.x] += rk[threadIdx.x + s]; }
    __syncthreads();
  }
  float invq = rsqrtf(rq[0] / (float)DMODEL + EPS_);
  float invk = rsqrtf(rk[0] / (float)DMODEL + EPS_);
  const int half = HDIM / 2;
  for (int p = threadIdx.x; p < NHEAD * half; p += 256) {
    int hh = p / half, i = p % half;
    size_t i1 = base + (size_t)hh * HDIM + i;
    size_t i2 = i1 + half;
    float freq = __expf(-((float)i / (float)half) * 9.210340371976184f); // 10000^-(i/half)
    float ang  = (float)t * freq;
    float c = __cosf(ang), s = __sinf(ang);
    float a1 = q[i1] * invq * qw[hh * HDIM + i];
    float a2 = q[i2] * invq * qw[hh * HDIM + half + i];
    q[i1] = a1 * c - a2 * s;
    q[i2] = a1 * s + a2 * c;
    float b1 = k[i1] * invk * kw[hh * HDIM + i];
    float b2 = k[i2] * invk * kw[hh * HDIM + half + i];
    k[i1] = b1 * c - b2 * s;
    k[i2] = b1 * s + b2 * c;
  }
}

// ---------------------------------------------------------------------------
// Flash attention: one wave32 per (16-query tile, head). Scores + P*V via WMMA.
// ---------------------------------------------------------------------------
__global__ __launch_bounds__(32)
void attn_kernel(const float* __restrict__ Q, const float* __restrict__ Kb,
                 const float* __restrict__ V, float* __restrict__ O)
{
  __shared__ float  st[16 * 32];
  __shared__ __bf16 pt[16 * 32];
  __shared__ float  fac[16];
  __shared__ float  lsum[16];

  int lane = threadIdx.x;
  int q0   = blockIdx.x * 16;
  int h    = blockIdx.y;
  int mrow = lane & 15;
  int kb   = (lane < 16) ? 0 : 8;
  size_t hoff = (size_t)h * HDIM;

  // Q fragments (4 x 16x32 over HDIM=128), pre-scaled by softmax scale.
  // Each (c,g) group is 8 consecutive floats -> two float4 loads.
  v16bf qf[4];
#pragma unroll
  for (int c = 0; c < 4; ++c) {
#pragma unroll
    for (int g = 0; g < 2; ++g) {
      int d0 = c * 32 + g * 16 + kb;
      const float4* qp4 = (const float4*)(Q + (size_t)(q0 + mrow) * DMODEL + hoff + d0);
      float4 u = qp4[0], v = qp4[1];
      qf[c][8*g+0] = (__bf16)(u.x * SCALE_); qf[c][8*g+1] = (__bf16)(u.y * SCALE_);
      qf[c][8*g+2] = (__bf16)(u.z * SCALE_); qf[c][8*g+3] = (__bf16)(u.w * SCALE_);
      qf[c][8*g+4] = (__bf16)(v.x * SCALE_); qf[c][8*g+5] = (__bf16)(v.y * SCALE_);
      qf[c][8*g+6] = (__bf16)(v.z * SCALE_); qf[c][8*g+7] = (__bf16)(v.w * SCALE_);
    }
  }

  v8f oacc[8];
#pragma unroll
  for (int tt = 0; tt < 8; ++tt)
#pragma unroll
    for (int r = 0; r < 8; ++r) oacc[tt][r] = 0.f;

  float m_i = -1e30f, l_i = 0.f;   // per-row stats (valid on lanes 0-15)

  for (int kk0 = 0; kk0 <= q0; kk0 += 32) {
    // ---- scores: two 16x16 tiles for 32 keys ----
#pragma unroll
    for (int hh = 0; hh < 2; ++hh) {
      v8f sacc;
#pragma unroll
      for (int r = 0; r < 8; ++r) sacc[r] = 0.f;
      int key = kk0 + hh * 16 + mrow;
#pragma unroll
      for (int c = 0; c < 4; ++c) {
        v16bf kf;
#pragma unroll
        for (int g = 0; g < 2; ++g) {
          int d0 = c * 32 + g * 16 + kb;
          const float4* kp4 = (const float4*)(Kb + (size_t)key * DMODEL + hoff + d0);
          float4 u = kp4[0], v = kp4[1];
          kf[8*g+0] = (__bf16)u.x; kf[8*g+1] = (__bf16)u.y;
          kf[8*g+2] = (__bf16)u.z; kf[8*g+3] = (__bf16)u.w;
          kf[8*g+4] = (__bf16)v.x; kf[8*g+5] = (__bf16)v.y;
          kf[8*g+6] = (__bf16)v.z; kf[8*g+7] = (__bf16)v.w;
        }
        sacc = wmma_bf16(qf[c], kf, sacc);
      }
#pragma unroll
      for (int r = 0; r < 8; ++r) {
        int row = (lane < 16) ? r : r + 8;
        st[row * 32 + hh * 16 + mrow] = sacc[r];
      }
    }
    __syncthreads();

    // ---- online softmax (lane == row, lanes 0-15) ----
    if (lane < 16) {
      int row = lane, qidx = q0 + row;
      float mx = m_i;
      float vals[32];
#pragma unroll
      for (int c2 = 0; c2 < 32; ++c2) {
        int key = kk0 + c2;
        float vv = (key <= qidx) ? st[row * 32 + c2] : -1e30f;
        vals[c2] = vv;
        mx = fmaxf(mx, vv);
      }
      float ssum = 0.f;
#pragma unroll
      for (int c2 = 0; c2 < 32; ++c2) {
        float pp = __expf(vals[c2] - mx);
        ssum += pp;
        pt[row * 32 + c2] = (__bf16)pp;
      }
      float f2 = __expf(m_i - mx);
      l_i = l_i * f2 + ssum;
      m_i = mx;
      fac[row] = f2;
    }
    __syncthreads();

    // ---- rescale accumulators, then O += P * V ----
#pragma unroll
    for (int tt = 0; tt < 8; ++tt)
#pragma unroll
      for (int r = 0; r < 8; ++r) {
        int row = (lane < 16) ? r : r + 8;
        oacc[tt][r] *= fac[row];
      }

    v16bf pf = load_frag(pt, 32, lane);
#pragma unroll
    for (int tt = 0; tt < 8; ++tt) {
      v16bf vf;
      int dcol = tt * 16 + mrow;
#pragma unroll
      for (int g = 0; g < 2; ++g)
#pragma unroll
        for (int j = 0; j < 4; ++j) {
          int kk = g * 16 + kb + 2 * j;
          int i  = g * 4 + j;
          vf[2 * i]     = (__bf16)V[(size_t)(kk0 + kk)     * DMODEL + hoff + dcol];
          vf[2 * i + 1] = (__bf16)V[(size_t)(kk0 + kk + 1) * DMODEL + hoff + dcol];
        }
      oacc[tt] = wmma_bf16(pf, vf, oacc[tt]);
    }
    __syncthreads();
  }

  if (lane < 16) lsum[lane] = l_i;
  __syncthreads();
#pragma unroll
  for (int tt = 0; tt < 8; ++tt)
#pragma unroll
    for (int r = 0; r < 8; ++r) {
      int row = (lane < 16) ? r : r + 8;
      O[(size_t)(q0 + row) * DMODEL + hoff + tt * 16 + mrow] = oacc[tt][r] / lsum[row];
    }
}

// ---------------------------------------------------------------------------
// Router softmax + top-8 + per-expert counting
// ---------------------------------------------------------------------------
__global__ void zero_counts_kernel(int* __restrict__ counts) {
  if (threadIdx.x < NEXP) counts[threadIdx.x] = 0;
}

__global__ void router_topk_kernel(const float* __restrict__ logits,
                                   float* __restrict__ tw, int* __restrict__ tidx,
                                   int* __restrict__ counts)
{
  int t = blockIdx.x * blockDim.x + threadIdx.x;
  if (t >= SEQ) return;
  const float* lg = logits + (size_t)t * NEXP;
  float mx = -1e30f;
  for (int e = 0; e < NEXP; ++e) mx = fmaxf(mx, lg[e]);
  float pr[NEXP];
  float sum = 0.f;
  for (int e = 0; e < NEXP; ++e) { pr[e] = __expf(lg[e] - mx); sum += pr[e]; }
  float inv = 1.f / sum;
  for (int kk = 0; kk < TOPK; ++kk) {
    int best = 0; float bv = -1.f;
    for (int e = 0; e < NEXP; ++e)
      if (pr[e] > bv) { bv = pr[e]; best = e; }
    tw[t * TOPK + kk]   = bv * inv;
    tidx[t * TOPK + kk] = best;
    pr[best] = -2.f;
    atomicAdd(&counts[best], 1);
  }
}

__global__ void offsets_kernel(const int* __restrict__ counts, int* __restrict__ off,
                               int* __restrict__ fill)
{
  if (threadIdx.x < NEXP) fill[threadIdx.x] = 0;
  if (threadIdx.x == 0) {
    int a = 0;
    for (int e = 0; e < NEXP; ++e) { off[e] = a; a += counts[e]; }
    off[NEXP] = a;
  }
}

__global__ void scatter_kernel(const int* __restrict__ tidx, const float* __restrict__ tw,
                               const int* __restrict__ off, int* __restrict__ fill,
                               int* __restrict__ tlist, float* __restrict__ wlist)
{
  int i = blockIdx.x * blockDim.x + threadIdx.x;
  if (i >= SEQ * TOPK) return;
  int e   = tidx[i];
  int pos = off[e] + atomicAdd(&fill[e], 1);
  tlist[pos] = i / TOPK;
  wlist[pos] = tw[i];
}

// h = silu(gate) * up * routing_weight, per gathered row
__global__ void silu_kernel(const float* __restrict__ gu, const float* __restrict__ wlist,
                            const int* __restrict__ off, float* __restrict__ hbuf)
{
  int i = blockIdx.x * blockDim.x + threadIdx.x;
  int r = i / MDIM, mcol = i % MDIM;
  if (r >= off[NEXP]) return;
  float g = gu[(size_t)r * (2 * MDIM) + mcol];
  float u = gu[(size_t)r * (2 * MDIM) + MDIM + mcol];
  float s = g / (1.f + __expf(-g));
  hbuf[(size_t)r * MDIM + mcol] = s * u * wlist[r];
}

// ---------------------------------------------------------------------------
extern "C" void kernel_launch(void* const* d_in, const int* in_sizes, int n_in,
                              void* d_out, int out_size, void* d_ws, size_t ws_size,
                              hipStream_t stream)
{
  (void)in_sizes; (void)n_in; (void)out_size; (void)ws_size;
  const int*   ids     = (const int*)  d_in[0];
  const float* embed_w = (const float*)d_in[1];
  const float* ln1_w   = (const float*)d_in[2];
  const float* q_w     = (const float*)d_in[3];
  const float* k_w     = (const float*)d_in[4];
  const float* v_w     = (const float*)d_in[5];
  const float* qn_w    = (const float*)d_in[6];
  const float* kn_w    = (const float*)d_in[7];
  const float* o_w     = (const float*)d_in[8];
  const float* ln2_w   = (const float*)d_in[9];
  const float* gate_w  = (const float*)d_in[10];
  const float* gup_w   = (const float*)d_in[11];
  const float* down_w  = (const float*)d_in[12];
  const float* norm_w  = (const float*)d_in[13];
  const float* lm_w    = (const float*)d_in[14];
  float* out = (float*)d_out;

  char* p = (char*)d_ws;
  auto carve = [&](size_t n) { void* r = (void*)p; p += (n + 255) & ~(size_t)255; return r; };
  float* x      = (float*)carve(sizeof(float) * SEQ * DMODEL);
  float* xn     = (float*)carve(sizeof(float) * SEQ * DMODEL);
  float* qb     = (float*)carve(sizeof(float) * SEQ * DMODEL);
  float* kbuf   = (float*)carve(sizeof(float) * SEQ * DMODEL);
  float* vb     = (float*)carve(sizeof(float) * SEQ * DMODEL);
  float* ctx    = (float*)carve(sizeof(float) * SEQ * DMODEL);
  float* x2     = (float*)carve(sizeof(float) * SEQ * DMODEL);
  float* x3     = (float*)carve(sizeof(float) * SEQ * DMODEL);
  float* hlast  = (float*)carve(sizeof(float) * SEQ * DMODEL);
  float* logits = (float*)carve(sizeof(float) * SEQ * NEXP);
  float* tw     = (float*)carve(sizeof(float) * SEQ * TOPK);
  float* wlist  = (float*)carve(sizeof(float) * SEQ * TOPK);
  float* gu     = (float*)carve(sizeof(float) * (size_t)SEQ * TOPK * 2 * MDIM);
  float* hbuf   = (float*)carve(sizeof(float) * (size_t)SEQ * TOPK * MDIM);
  int* tidx   = (int*)carve(sizeof(int) * SEQ * TOPK);
  int* counts = (int*)carve(sizeof(int) * NEXP);
  int* eoff   = (int*)carve(sizeof(int) * (NEXP + 1));
  int* fill   = (int*)carve(sizeof(int) * NEXP);
  int* tlist  = (int*)carve(sizeof(int) * SEQ * TOPK);

  // 1. embed + RMS
  embed_rms_kernel<<<SEQ, 256, 0, stream>>>(ids, embed_w, ln1_w, x, xn);

  // 2. Q/K/V projections
  dim3 g_dd(DMODEL / BN, SEQ / BM);
  gemm_wmma_kernel<<<g_dd, 256, 0, stream>>>(xn, DMODEL, q_w, DMODEL, qb, DMODEL,
      SEQ, DMODEL, DMODEL, nullptr, nullptr, nullptr, nullptr, 0);
  gemm_wmma_kernel<<<g_dd, 256, 0, stream>>>(xn, DMODEL, k_w, DMODEL, kbuf, DMODEL,
      SEQ, DMODEL, DMODEL, nullptr, nullptr, nullptr, nullptr, 0);
  gemm_wmma_kernel<<<g_dd, 256, 0, stream>>>(xn, DMODEL, v_w, DMODEL, vb, DMODEL,
      SEQ, DMODEL, DMODEL, nullptr, nullptr, nullptr, nullptr, 0);

  // 3. q/k RMS + RoPE
  qk_rms_rope_kernel<<<SEQ, 256, 0, stream>>>(qb, kbuf, qn_w, kn_w);

  // 4. causal flash attention
  attn_kernel<<<dim3(SEQ / 16, NHEAD), 32, 0, stream>>>(qb, kbuf, vb, ctx);

  // 5. output projection + residual
  gemm_wmma_kernel<<<g_dd, 256, 0, stream>>>(ctx, DMODEL, o_w, DMODEL, x2, DMODEL,
      SEQ, DMODEL, DMODEL, x, nullptr, nullptr, nullptr, 0);

  // 6. pre-MoE RMS
  rms_kernel<<<SEQ, 256, 0, stream>>>(x2, ln2_w, x3);

  // 7. router logits (N=64)
  gemm_wmma_kernel<<<dim3(1, SEQ / BM), 256, 0, stream>>>(x3, DMODEL, gate_w, DMODEL,
      logits, NEXP, SEQ, NEXP, DMODEL, nullptr, nullptr, nullptr, nullptr, 0);

  // 8. top-8 routing + expert bucketing
  zero_counts_kernel<<<1, 64, 0, stream>>>(counts);
  router_topk_kernel<<<(SEQ + 255) / 256, 256, 0, stream>>>(logits, tw, tidx, counts);
  offsets_kernel<<<1, 64, 0, stream>>>(counts, eoff, fill);
  scatter_kernel<<<(SEQ * TOPK + 255) / 256, 256, 0, stream>>>(tidx, tw, eoff, fill,
      tlist, wlist);

  // 9. MoE gate_up: gathered rows per expert -> gu[r, 2M]
  gemm_wmma_kernel<<<dim3(2 * MDIM / BN, (SEQ * TOPK) / BM, NEXP), 256, 0, stream>>>(
      x3, DMODEL, gup_w, DMODEL, gu, 2 * MDIM, SEQ * TOPK, 2 * MDIM, DMODEL,
      nullptr, tlist, nullptr, eoff, (long long)(2 * MDIM) * DMODEL);

  // 10. SiLU * up * routing weight
  silu_kernel<<<(SEQ * TOPK * MDIM + 255) / 256, 256, 0, stream>>>(gu, wlist, eoff, hbuf);

  // 11. MoE down: scatter-add into residual stream x2
  gemm_wmma_kernel<<<dim3(DMODEL / BN, (SEQ * TOPK) / BM, NEXP), 256, 0, stream>>>(
      hbuf, MDIM, down_w, MDIM, x2, DMODEL, SEQ * TOPK, DMODEL, MDIM,
      nullptr, nullptr, tlist, eoff, (long long)DMODEL * MDIM);

  // 12. final RMS
  rms_kernel<<<SEQ, 256, 0, stream>>>(x2, norm_w, hlast);

  // 13. LM head (N = 50304 = 393*128)
  gemm_wmma_kernel<<<dim3(VOCAB / BN, SEQ / BM), 256, 0, stream>>>(hlast, DMODEL,
      lm_w, DMODEL, out, VOCAB, SEQ, VOCAB, DMODEL,
      nullptr, nullptr, nullptr, nullptr, 0);
}
